// ContextualAttention_Enhance_6579889898012
// MI455X (gfx1250) — matrix operand order repaired and checked by reading, not compiled, not run on hardware
//
#include <hip/hip_runtime.h>

typedef __bf16 bf16;
typedef __attribute__((ext_vector_type(16))) __bf16 v16bf;
typedef __attribute__((ext_vector_type(8)))  __bf16 v8bf;
typedef __attribute__((ext_vector_type(8)))  float  v8f;

#define BN   4
#define HH   96
#define WW   96
#define CIN_ 64
#define CI_  16
#define L1_  576     // 24*24 query patches
#define L2_  9216    // 96*96 key positions
#define PP_  784     // CI*7*7
#define KP1_ 800     // K padded for GEMM1 (25*32)
#define MP2_ 832     // M padded for GEMM2 (13*64)
#define NP2_ 640     // N padded for GEMM2 (5*128)
#define SCALE_ 10.0f
#define TOPK_ 500

// ---------------------------------------------------------------- zero fill
__global__ void k_zero(uint4* __restrict__ p, long n) {
  long i = (long)blockIdx.x * blockDim.x + threadIdx.x;
  long stride = (long)gridDim.x * blockDim.x;
  uint4 z; z.x = 0u; z.y = 0u; z.z = 0u; z.w = 0u;
  for (; i < n; i += stride) p[i] = z;
}

// ------------------------------------------- fused 3x conv1x1 (64 -> 3*16)
// writes padded bf16 planes: b1pad (pad 1, queries), b2pad/b3pad (pad 3)
__global__ void k_conv3(const float* __restrict__ x,
                        const float* __restrict__ gw, const float* __restrict__ gb,
                        const float* __restrict__ tw, const float* __restrict__ tb,
                        const float* __restrict__ pw, const float* __restrict__ pb,
                        bf16* __restrict__ b1pad, bf16* __restrict__ b2pad,
                        bf16* __restrict__ b3pad) {
  int t = blockIdx.x * blockDim.x + threadIdx.x;
  if (t >= BN * HH * WW) return;
  int b = t / (HH * WW);
  int r = t % (HH * WW);
  int h = r / WW, w = r % WW;
  float ag[CI_] = {}, at[CI_] = {}, ap[CI_] = {};
  const float* xb = x + (long)b * CIN_ * HH * WW + r;
  for (int c = 0; c < CIN_; ++c) {
    float xv = xb[(long)c * HH * WW];
    for (int o = 0; o < CI_; ++o) {
      ag[o] += gw[o * CIN_ + c] * xv;
      at[o] += tw[o * CIN_ + c] * xv;
      ap[o] += pw[o * CIN_ + c] * xv;
    }
  }
  for (int o = 0; o < CI_; ++o) {
    b1pad[(((long)b * CI_ + o) * 100 + (h + 1)) * 100 + (w + 1)] = (bf16)(ag[o] + gb[o]);
    b2pad[(((long)b * CI_ + o) * 102 + (h + 3)) * 102 + (w + 3)] = (bf16)(at[o] + tb[o]);
    b3pad[(((long)b * CI_ + o) * 102 + (h + 3)) * 102 + (w + 3)] = (bf16)(ap[o] + pb[o]);
  }
}

// -------------------------------------------------- im2col for queries (wi)
__global__ void k_im2col_q(const bf16* __restrict__ b1pad, bf16* __restrict__ wi) {
  long t = (long)blockIdx.x * blockDim.x + threadIdx.x;
  long total = (long)BN * L1_ * PP_;
  if (t >= total) return;
  int p = (int)(t % PP_);
  long r = t / PP_;
  int l = (int)(r % L1_);
  int b = (int)(r / L1_);
  int c = p / 49, q = p % 49, ki = q / 7, kj = q % 7;
  int oh = l / 24, ow = l % 24;
  wi[((long)b * L1_ + l) * KP1_ + p] =
      b1pad[(((long)b * CI_ + c) * 100 + oh * 4 + ki) * 100 + ow * 4 + kj];
}

// --------------------------------------------- im2col for keys/values (xi/pi)
__global__ void k_im2col_kv(const bf16* __restrict__ b3pad, const bf16* __restrict__ b2pad,
                            bf16* __restrict__ xi, bf16* __restrict__ pi) {
  long t = (long)blockIdx.x * blockDim.x + threadIdx.x;
  long total = (long)BN * PP_ * L2_;
  if (t >= total) return;
  int m = (int)(t % L2_);
  long r = t / L2_;
  int p = (int)(r % PP_);
  int b = (int)(r / PP_);
  int c = p / 49, q = p % 49, ki = q / 7, kj = q % 7;
  int h = m / WW, w = m % WW;
  long src = (((long)b * CI_ + c) * 102 + h + ki) * 102 + w + kj;
  xi[((long)b * KP1_ + p) * L2_ + m] = b3pad[src];
  pi[((long)b * MP2_ + p) * L2_ + m] = b2pad[src];
}

// ------------------------------------------------- generic bf16 WMMA GEMM
// C[M x N] (f32, row-major, ldc) = A[M x K] (bf16, lda) * B[K x N] (bf16, ldb)
// Workgroup tile 64x128, 8 waves, wave tile 16x64 (4 wmma accumulators).
__global__ __launch_bounds__(256)
void k_gemm(const bf16* __restrict__ A, long lda, long sA,
            const bf16* __restrict__ Bm, long ldb, long sB,
            float* __restrict__ C, long ldc, long sC,
            int Kpad, int Mreal, int Nreal) {
  __shared__ __align__(64) bf16 As[64][32];   // [m][k]
  __shared__ __align__(64) bf16 Bs[128][32];  // [n][k] (transposed stage)

  int b = blockIdx.z;
  const bf16* Ab = A + (long)b * sA;
  const bf16* Bb = Bm + (long)b * sB;
  float* Cb = C + (long)b * sC;
  int Mtile = blockIdx.y * 64;
  int Ntile = blockIdx.x * 128;

  unsigned tid = threadIdx.x;
  unsigned lane = tid & 31u, wave = tid >> 5;
  unsigned wm = wave & 3u, wn = wave >> 2;    // 4 (M) x 2 (N) waves
  unsigned hsel = lane >> 4, lcol = lane & 15u;

  v8f acc[4] = {};

  int arow = tid >> 2;           // 0..63
  int acol = (tid & 3) * 8;      // 0,8,16,24
  int bkrow = tid >> 3;          // 0..31
  int bn0 = (tid & 7) * 16;      // 0..112

  for (int k0 = 0; k0 < Kpad; k0 += 32) {
    __syncthreads();
    // stage A tile (64x32), 16B vector per thread
    *(v8bf*)&As[arow][acol] = *(const v8bf*)(Ab + (long)(Mtile + arow) * lda + k0 + acol);
    // stage B tile (32x128) transposed into Bs[n][k]
    v16bf btmp = *(const v16bf*)(Bb + (long)(k0 + bkrow) * ldb + Ntile + bn0);
    for (int i = 0; i < 16; ++i) Bs[bn0 + i][bkrow] = btmp[i];
    __syncthreads();

    // A fragment: row = lcol of wave's M-subtile; contiguous K run of 16 at hsel*16
    const bf16* ar = &As[wm * 16 + lcol][0];
    v16bf afrag = *(const v16bf*)(ar + hsel * 16);
    for (int f = 0; f < 4; ++f) {
      // B fragment: column = lcol, contiguous K run of 16 at hsel*16
      const bf16* br = &Bs[wn * 64 + f * 16 + lcol][0];
      v16bf bfrag = *(const v16bf*)(br + hsel * 16);
      acc[f] = __builtin_amdgcn_wmma_f32_16x16x32_bf16(
          false, afrag, false, bfrag, (short)0, acc[f], false, false);
    }
  }

  for (int f = 0; f < 4; ++f) {
    int n = Ntile + (int)(wn * 64 + f * 16 + lcol);
    if (n >= Nreal) continue;
    for (int r = 0; r < 8; ++r) {
      int m = Mtile + (int)(wm * 16 + hsel * 8) + r;
      if (m < Mreal) Cb[(long)m * ldc + n] = acc[f][r];
    }
  }
}

// ----------------------------- top-k(500) mask + masked softmax, write attn^T
__global__ __launch_bounds__(256)
void k_topk_softmax(const float* __restrict__ score, bf16* __restrict__ attnT) {
  __shared__ float row[L2_];
  __shared__ float redf[256];
  __shared__ int redi[256];
  int blk = blockIdx.x;          // b * L1 + l
  int b = blk / L1_, l = blk % L1_;
  int tid = threadIdx.x;
  const float* src = score + ((long)b * L1_ + l) * L2_;
  for (int i = tid; i < L2_; i += 256) row[i] = src[i];
  __syncthreads();

  // exact bisection on the monotonic uint mapping: largest thr with count>=TOPK
  unsigned lo = 0u, hi = 0xFFFFFFFFu;
  for (int it = 0; it < 32 && lo < hi; ++it) {
    unsigned d = hi - lo;
    unsigned mid = lo + (d >> 1) + (d & 1u);
    int c = 0;
    for (int i = tid; i < L2_; i += 256) {
      unsigned bits = __float_as_uint(row[i]);
      unsigned u = bits ^ (unsigned)(((int)bits >> 31) | (int)0x80000000);
      c += (u >= mid) ? 1 : 0;
    }
    redi[tid] = c; __syncthreads();
    for (int s = 128; s > 0; s >>= 1) {
      if (tid < s) redi[tid] += redi[tid + s];
      __syncthreads();
    }
    int total = redi[0]; __syncthreads();
    if (total >= TOPK_) lo = mid; else hi = mid - 1;
  }
  unsigned thr = lo;

  // max over score*mask*SCALE (masked entries are 0, per reference)
  float mx = -3.4e38f;
  for (int i = tid; i < L2_; i += 256) {
    unsigned bits = __float_as_uint(row[i]);
    unsigned u = bits ^ (unsigned)(((int)bits >> 31) | (int)0x80000000);
    float v = (u >= thr) ? row[i] * SCALE_ : 0.0f;
    mx = fmaxf(mx, v);
  }
  redf[tid] = mx; __syncthreads();
  for (int s = 128; s > 0; s >>= 1) {
    if (tid < s) redf[tid] = fmaxf(redf[tid], redf[tid + s]);
    __syncthreads();
  }
  float M = redf[0]; __syncthreads();

  float sum = 0.0f;
  for (int i = tid; i < L2_; i += 256) {
    unsigned bits = __float_as_uint(row[i]);
    unsigned u = bits ^ (unsigned)(((int)bits >> 31) | (int)0x80000000);
    float v = (u >= thr) ? row[i] * SCALE_ : 0.0f;
    sum += __expf(v - M);
  }
  redf[tid] = sum; __syncthreads();
  for (int s = 128; s > 0; s >>= 1) {
    if (tid < s) redf[tid] += redf[tid + s];
    __syncthreads();
  }
  float inv = 1.0f / redf[0];

  for (int i = tid; i < L2_; i += 256) {
    unsigned bits = __float_as_uint(row[i]);
    unsigned u = bits ^ (unsigned)(((int)bits >> 31) | (int)0x80000000);
    float a = (u >= thr) ? __expf(row[i] * SCALE_ - M) * inv : 0.0f;
    attnT[((long)b * L2_ + i) * NP2_ + l] = (bf16)a;
  }
}

// ----------------------------- fold (overlap-add gather) + count normalize
__global__ void k_fold(const float* __restrict__ agg, float* __restrict__ z) {
  int t = blockIdx.x * blockDim.x + threadIdx.x;
  if (t >= BN * CI_ * HH * WW) return;
  int w = t % WW;
  int r = t / WW;
  int h = r % HH; r /= HH;
  int c = r % CI_;
  int b = r / CI_;
  int th = h - 3;
  int ohlo = th <= 0 ? 0 : (th + 3) >> 2;
  int ohhi = (h + 3) >> 2; if (ohhi > 23) ohhi = 23;
  int tw = w - 3;
  int owlo = tw <= 0 ? 0 : (tw + 3) >> 2;
  int owhi = (w + 3) >> 2; if (owhi > 23) owhi = 23;
  const float* ab = agg + (long)b * PP_ * L1_;
  float s = 0.0f;
  for (int oh = ohlo; oh <= ohhi; ++oh) {
    int ki = h + 3 - 4 * oh;
    for (int ow = owlo; ow <= owhi; ++ow) {
      int kj = w + 3 - 4 * ow;
      s += ab[(long)(c * 49 + ki * 7 + kj) * L1_ + oh * 24 + ow];
    }
  }
  float cnt = (float)((ohhi - ohlo + 1) * (owhi - owlo + 1));
  z[t] = s / cnt;
}

// ------------------------------------- final conv1x1 (16 -> 64) + residual
__global__ void k_out(const float* __restrict__ z, const float* __restrict__ Ww,
                      const float* __restrict__ Wb, const float* __restrict__ x,
                      float* __restrict__ out) {
  int t = blockIdx.x * blockDim.x + threadIdx.x;
  if (t >= BN * HH * WW) return;
  int b = t / (HH * WW);
  int r = t % (HH * WW);
  float zv[CI_];
  for (int c = 0; c < CI_; ++c) zv[c] = z[((long)b * CI_ + c) * HH * WW + r];
  for (int o = 0; o < CIN_; ++o) {
    float acc = Wb[o];
    for (int c = 0; c < CI_; ++c) acc += Ww[o * CI_ + c] * zv[c];
    long idx = ((long)b * CIN_ + o) * HH * WW + r;
    out[idx] = x[idx] + acc;
  }
}

extern "C" void kernel_launch(void* const* d_in, const int* in_sizes, int n_in,
                              void* d_out, int out_size, void* d_ws, size_t ws_size,
                              hipStream_t stream) {
  const float* x  = (const float*)d_in[0];
  const float* gw = (const float*)d_in[1];
  const float* gb = (const float*)d_in[2];
  const float* tw = (const float*)d_in[3];
  const float* tb = (const float*)d_in[4];
  const float* pw = (const float*)d_in[5];
  const float* pb = (const float*)d_in[6];
  const float* Ww = (const float*)d_in[7];
  const float* Wb = (const float*)d_in[8];
  float* out = (float*)d_out;

  char* ws = (char*)d_ws;
  size_t off = 0;
  auto alloc = [&](size_t bytes) -> size_t {
    size_t r = off;
    off = (off + bytes + 255) & ~(size_t)255;
    return r;
  };
  size_t o_b1 = alloc((size_t)BN * CI_ * 100 * 100 * 2);
  size_t o_b2 = alloc((size_t)BN * CI_ * 102 * 102 * 2);
  size_t o_b3 = alloc((size_t)BN * CI_ * 102 * 102 * 2);
  size_t o_wi = alloc((size_t)BN * L1_ * KP1_ * 2);
  size_t o_xi = alloc((size_t)BN * KP1_ * L2_ * 2);
  size_t o_pi = alloc((size_t)BN * MP2_ * L2_ * 2);
  size_t o_sc = alloc((size_t)BN * L1_ * L2_ * 4);
  size_t o_at = alloc((size_t)BN * L2_ * NP2_ * 2);
  size_t o_ag = alloc((size_t)BN * PP_ * L1_ * 4);
  size_t o_z  = alloc((size_t)BN * CI_ * HH * WW * 4);
  (void)o_z; (void)ws_size; (void)in_sizes; (void)n_in; (void)out_size;

  bf16* b1pad = (bf16*)(ws + o_b1);
  bf16* b2pad = (bf16*)(ws + o_b2);
  bf16* b3pad = (bf16*)(ws + o_b3);
  bf16* wi    = (bf16*)(ws + o_wi);
  bf16* xi    = (bf16*)(ws + o_xi);
  bf16* pi    = (bf16*)(ws + o_pi);
  float* score = (float*)(ws + o_sc);
  bf16* attnT = (bf16*)(ws + o_at);
  float* agg  = (float*)(ws + o_ag);
  float* zbuf = (float*)(ws + o_z);

  // zero padded-staging region (b1pad..pi) and attnT padding columns
  k_zero<<<2048, 256, 0, stream>>>((uint4*)ws, (long)(o_sc / 16));
  k_zero<<<2048, 256, 0, stream>>>((uint4*)(ws + o_at),
                                   (long)(((size_t)BN * L2_ * NP2_ * 2) / 16));

  k_conv3<<<(BN * HH * WW + 255) / 256, 256, 0, stream>>>(
      x, gw, gb, tw, tb, pw, pb, b1pad, b2pad, b3pad);

  long nq = (long)BN * L1_ * PP_;
  k_im2col_q<<<(unsigned)((nq + 255) / 256), 256, 0, stream>>>(b1pad, wi);
  long nkv = (long)BN * PP_ * L2_;
  k_im2col_kv<<<(unsigned)((nkv + 255) / 256), 256, 0, stream>>>(b3pad, b2pad, xi, pi);

  // GEMM1: score[576 x 9216] = wi[576 x 800] * xi[800 x 9216]
  dim3 g1(L2_ / 128, L1_ / 64, BN);
  k_gemm<<<g1, 256, 0, stream>>>(wi, KP1_, (long)L1_ * KP1_,
                                 xi, L2_, (long)KP1_ * L2_,
                                 score, L2_, (long)L1_ * L2_,
                                 KP1_, L1_, L2_);

  k_topk_softmax<<<BN * L1_, 256, 0, stream>>>(score, attnT);

  // GEMM2: agg[784 x 576] = pi[784 x 9216] * attnT[9216 x 576]  (padded 832x640)
  dim3 g2(NP2_ / 128, MP2_ / 64, BN);
  k_gemm<<<g2, 256, 0, stream>>>(pi, L2_, (long)MP2_ * L2_,
                                 attnT, NP2_, (long)L2_ * NP2_,
                                 agg, L1_, (long)PP_ * L1_,
                                 L2_, PP_, L1_);

  k_fold<<<(BN * CI_ * HH * WW + 255) / 256, 256, 0, stream>>>(agg, zbuf);
  k_out<<<(BN * HH * WW + 255) / 256, 256, 0, stream>>>(zbuf, Ww, Wb, x, out);
}
